// SpatialAlignCrossAttentionCosFormer_65841848648188
// MI455X (gfx1250) — matrix operand order repaired and checked by reading, not compiled, not run on hardware
//
#include <hip/hip_runtime.h>
#include <hip/hip_bf16.h>

// ---- CDNA5 (gfx1250) wave32 WMMA implementation ----
typedef __attribute__((ext_vector_type(16))) __bf16 v16bf;
typedef __attribute__((ext_vector_type(8)))  __bf16 v8bf;
typedef __attribute__((ext_vector_type(8)))  float  v8f;

#define BSZ   16
#define C1    256
#define C2    512
#define HW3   6400   /* 80*80 */
#define W4D   40
#define N4    1600   /* 40*40 (tgt == src) */
#define TTOK  25600  /* BSZ*N4 */
#define NHEAD 8
#define HDIM  32

__device__ __forceinline__ float san(float x) {
  if (__builtin_isnan(x)) return 0.f;
  if (__builtin_isinf(x)) return x > 0.f ? 1e4f : -1e4f;
  return x;
}
__device__ __forceinline__ __bf16 f2bf(float f) {
  unsigned u = __float_as_uint(f);
  u += 0x7fffu + ((u >> 16) & 1u);           // round-to-nearest-even
  unsigned short h = (unsigned short)(u >> 16);
  return __builtin_bit_cast(__bf16, h);
}

// ---------------- pos embedding table: pos[p*256 + c] ----------------
__global__ void k_pos(float* __restrict__ pos) {
  int i = blockIdx.x * 256 + threadIdx.x;
  if (i >= N4 * C1) return;
  int c = i & 255, p = i >> 8;
  int j = c & 63, r = c >> 6;
  float om = __expf(-(float)j * 0.14391156f);          // ln(10000)/64
  float g  = (r < 2) ? (float)(p / W4D) : (float)(p % W4D);
  float a  = g * om;
  pos[i] = (r & 1) ? __cosf(a) : __sinf(a);
}

// ---------------- fp32 -> bf16 weight convert ----------------
__global__ void k_cvt(const float* __restrict__ s, __bf16* __restrict__ d, int n) {
  int i = blockIdx.x * 256 + threadIdx.x;
  if (i < n) d[i] = f2bf(s[i]);
}

// ---------------- sanitize+avgpool2 a3 -> resid fp32 [t][c], a3q = bf16(a3d+pos) ----------------
__global__ void k_prep_a3(const float* __restrict__ a3, const float* __restrict__ pos,
                          float* __restrict__ resid, __bf16* __restrict__ a3q) {
  int i = blockIdx.x * 256 + threadIdx.x;
  if (i >= BSZ * C1 * N4) return;
  int p = i % N4; int c = (i / N4) % C1; int b = i / (N4 * C1);
  int y = p / W4D, x = p % W4D;
  const float* s = a3 + ((size_t)(b * C1 + c)) * HW3 + (2 * y) * 80 + 2 * x;
  float v = 0.25f * (san(s[0]) + san(s[1]) + san(s[80]) + san(s[81]));
  size_t t = (size_t)b * N4 + p;
  resid[t * C1 + c] = v;
  a3q[t * C1 + c]   = f2bf(v + pos[p * C1 + c]);
}

// ---------------- a4 (b,c,hw) -> token-major bf16 [t][512] (LDS tiled transpose) ----------------
__global__ void k_a4tok(const float* __restrict__ a4, __bf16* __restrict__ a4tok) {
  __shared__ float tile[32][33];
  int b = blockIdx.z, ct = blockIdx.y, pt = blockIdx.x;
  int lx = threadIdx.x & 31, ly = threadIdx.x >> 5;
#pragma unroll
  for (int r = 0; r < 4; ++r) {
    int c = ct * 32 + ly + r * 8, p = pt * 32 + lx;
    tile[ly + r * 8][lx] = san(a4[((size_t)(b * C2 + c)) * N4 + p]);
  }
  __syncthreads();
#pragma unroll
  for (int r = 0; r < 4; ++r) {
    int p = pt * 32 + ly + r * 8, c = ct * 32 + lx;
    a4tok[((size_t)b * N4 + p) * C2 + c] = f2bf(tile[lx][ly + r * 8]);
  }
}

// ---------------- WMMA GEMM: C[M,N] = A[M,K](bf16) * W[N,K]^T(bf16) + epilogue ----------------
// EPI 0: Cf = acc + bias                        (fp32 out)
// EPI 1: Cb = bf16((acc+bias)*rs*bn_g + bn_b + pos)   (conv+BN+posembed, N==256)
// EPI 3: Cb = bf16(gelu_exact(acc+bias))
template <int EPI>
__global__ void k_gemm(const __bf16* __restrict__ A, const __bf16* __restrict__ B,
                       const float* __restrict__ bias, float* __restrict__ Cf,
                       __bf16* __restrict__ Cb, const float* __restrict__ pos,
                       const float* __restrict__ bng, const float* __restrict__ bnb,
                       int M, int N, int K) {
  const int wid  = (blockIdx.x * blockDim.x + threadIdx.x) >> 5;
  const int lane = threadIdx.x & 31;
  const int tn4  = N >> 6;                       // 64-wide strips
  const int tm = wid / tn4, tn = wid % tn4;
  const int row0 = tm << 4, col0 = tn << 6;
  if (row0 >= M) return;
  const int m = lane & 15, g = lane >> 4;        // A row / B col in tile; K-half select
  v8f acc[4];
#pragma unroll
  for (int j = 0; j < 4; ++j) acc[j] = 0.f;
  const __bf16* Ap = A + (size_t)(row0 + m) * K + g * 16;
  const __bf16* Bp = B + (size_t)(col0 + m) * K + g * 16;
  for (int k0 = 0; k0 < K; k0 += 32) {
    __builtin_prefetch(Ap + k0 + 128, 0, 1);     // global_prefetch_b8
    // A fragment (16x32): lane holds K in [16g,16g+16) of row m
    v8bf lo = *(const v8bf*)(Ap + k0);
    v8bf hi = *(const v8bf*)(Ap + k0 + 8);
    v16bf af;
#pragma unroll
    for (int i = 0; i < 8; ++i) { af[i] = lo[i]; af[i + 8] = hi[i]; }
#pragma unroll
    for (int j = 0; j < 4; ++j) {
      // B fragment (32x16): lane holds K in [16g,16g+16) of W row (col0+16j+m)
      v16bf bf = *(const v16bf*)(Bp + (size_t)(j * 16) * K + k0);
      acc[j] = __builtin_amdgcn_wmma_f32_16x16x32_bf16(false, af, false, bf,
                                                       (short)0, acc[j], false, false);
    }
  }
#pragma unroll
  for (int j = 0; j < 4; ++j) {
    int col = col0 + j * 16 + m;
    float bz = bias[col];
#pragma unroll
    for (int r = 0; r < 8; ++r) {
      int row = row0 + r + 8 * g;                // D layout: vgpr r -> M = r + 8*(lane>=16)
      float v = acc[j][r] + bz;
      if constexpr (EPI == 0) {
        Cf[(size_t)row * N + col] = v;
      } else if constexpr (EPI == 1) {
        v = v * 0.99999500f * bng[col] + bnb[col];   // 1/sqrt(1+1e-5)
        v += pos[(row % N4) * C1 + col];
        Cb[(size_t)row * N + col] = f2bf(v);
      } else {
        float gl = 0.5f * v * (1.f + erff(v * 0.70710678f));
        Cb[(size_t)row * N + col] = f2bf(gl);
      }
    }
  }
}

// ---------------- wave-per-token LayerNorm (+relu / +residual / +bf16 copy) ----------------
// mode 0: Y = relu(LN(X))            mode 1: Y = R + LN(X), Yb = bf16(Y)   mode 2: Y = R + LN(X)
__global__ void k_ln(const float* __restrict__ X, const float* __restrict__ G,
                     const float* __restrict__ Bt, const float* __restrict__ R,
                     float* __restrict__ Y, __bf16* __restrict__ Yb, int mode) {
  int t = (blockIdx.x * blockDim.x + threadIdx.x) >> 5;
  int lane = threadIdx.x & 31;
  if (t >= TTOK) return;
  const float* x = X + (size_t)t * C1;
  float v[8], s = 0.f, s2 = 0.f;
#pragma unroll
  for (int i = 0; i < 8; ++i) { v[i] = x[lane + 32 * i]; s += v[i]; s2 += v[i] * v[i]; }
#pragma unroll
  for (int o = 16; o > 0; o >>= 1) { s += __shfl_xor(s, o, 32); s2 += __shfl_xor(s2, o, 32); }
  float mean = s * (1.f / C1);
  float var  = s2 * (1.f / C1) - mean * mean;
  float rstd = rsqrtf(var + 1e-5f);
#pragma unroll
  for (int i = 0; i < 8; ++i) {
    int c = lane + 32 * i;
    float y = (v[i] - mean) * rstd * G[c] + Bt[c];
    if (mode == 0) {
      Y[(size_t)t * C1 + c] = fmaxf(y, 0.f);
    } else {
      y = R[(size_t)t * C1 + c] + y;
      Y[(size_t)t * C1 + c] = y;
      if (mode == 1) Yb[(size_t)t * C1 + c] = f2bf(y);
    }
  }
}

// ---------------- cosFormer linear attention: one block per (b, head) ----------------
__global__ void k_attn(const float* __restrict__ Qf, const float* __restrict__ Kf,
                       const float* __restrict__ Vf, __bf16* __restrict__ attn) {
  __shared__ float KV[64 * 32];
  __shared__ float KS[64];
  int bh = blockIdx.x; int b = bh >> 3; int h = bh & 7;
  int e = threadIdx.x & 31;          // output feature within head
  int dg = threadIdx.x >> 5;         // 0..7, owns d = dg + 8j
  const float* Kb = Kf + ((size_t)b * N4) * C1 + h * HDIM;
  const float* Vb = Vf + ((size_t)b * N4) * C1 + h * HDIM;
  float acc[8], ks[8];
#pragma unroll
  for (int j = 0; j < 8; ++j) { acc[j] = 0.f; ks[j] = 0.f; }
  const float wstep = 1.5707963268f / (float)N4;
  for (int s = 0; s < N4; ++s) {
    float w = (float)(s + 1) * wstep;
    float sw = __sinf(w), cw = __cosf(w);
    const float* krow = Kb + (size_t)s * C1;
    float v = Vb[(size_t)s * C1 + e];
#pragma unroll
    for (int j = 0; j < 8; ++j) {
      int d = dg + 8 * j;
      float kv = krow[d & 31] * (d < 32 ? sw : cw);
      acc[j] += kv * v;
      ks[j]  += kv;
    }
  }
#pragma unroll
  for (int j = 0; j < 8; ++j) KV[(dg + 8 * j) * 32 + e] = acc[j];
  if (e == 0) {
#pragma unroll
    for (int j = 0; j < 8; ++j) KS[dg + 8 * j] = ks[j];
  }
  __syncthreads();
  const float* Qb = Qf + ((size_t)b * N4) * C1 + h * HDIM;
  __bf16* Ab = attn + ((size_t)b * N4) * C1 + h * HDIM;
  for (int l = dg; l < N4; l += 8) {
    float wl = (float)(l + 1) * wstep;
    float sl = __sinf(wl), cl = __cosf(wl);
    const float* qrow = Qb + (size_t)l * C1;
    float num = 0.f, den = 0.f;
#pragma unroll 4
    for (int dd = 0; dd < 32; ++dd) {
      float q = qrow[dd];
      float qs = q * sl, qc = q * cl;
      num += qs * KV[dd * 32 + e] + qc * KV[(dd + 32) * 32 + e];
      den += qs * KS[dd] + qc * KS[dd + 32];
    }
    float z = 1.f / fminf(fmaxf(den, 1e-4f), 10.f);
    Ab[(size_t)l * C1 + e] = f2bf(num * z);
  }
}

// ---------------- bilinear helpers / upsample kernels ----------------
__device__ __forceinline__ void bilin(int o, int S, int& i0, int& i1, float& f) {
  float s = (o + 0.5f) * 0.5f - 0.5f;
  float fl = floorf(s);
  f = s - fl;
  int i = (int)fl;
  i0 = i < 0 ? 0 : i;
  i1 = (i + 1) > S - 1 ? S - 1 : i + 1;
}

__global__ void k_up_a3(const float* __restrict__ out2, const float* __restrict__ a3,
                        float* __restrict__ dst) {
  size_t i = (size_t)blockIdx.x * 256 + threadIdx.x;
  if (i >= (size_t)BSZ * C1 * HW3) return;
  int ox = (int)(i % 80); int oy = (int)((i / 80) % 80);
  int c = (int)((i / HW3) % C1); int b = (int)(i / ((size_t)HW3 * C1));
  int y0, y1, x0, x1; float fy, fx;
  bilin(oy, W4D, y0, y1, fy); bilin(ox, W4D, x0, x1, fx);
  const float* base = out2 + (size_t)b * N4 * C1 + c;
  float v00 = base[(size_t)(y0 * W4D + x0) * C1];
  float v01 = base[(size_t)(y0 * W4D + x1) * C1];
  float v10 = base[(size_t)(y1 * W4D + x0) * C1];
  float v11 = base[(size_t)(y1 * W4D + x1) * C1];
  float v = (1.f - fy) * ((1.f - fx) * v00 + fx * v01) + fy * ((1.f - fx) * v10 + fx * v11);
  dst[i] = v * san(a3[i]);
}

__global__ void k_up_a4(const float* __restrict__ a4, float* __restrict__ dst) {
  size_t i = (size_t)blockIdx.x * 256 + threadIdx.x;
  if (i >= (size_t)BSZ * C2 * HW3) return;
  int ox = (int)(i % 80); int oy = (int)((i / 80) % 80);
  int c = (int)((i / HW3) % C2); int b = (int)(i / ((size_t)HW3 * C2));
  int y0, y1, x0, x1; float fy, fx;
  bilin(oy, W4D, y0, y1, fy); bilin(ox, W4D, x0, x1, fx);
  const float* base = a4 + ((size_t)(b * C2 + c)) * N4;
  float v00 = san(base[y0 * W4D + x0]);
  float v01 = san(base[y0 * W4D + x1]);
  float v10 = san(base[y1 * W4D + x0]);
  float v11 = san(base[y1 * W4D + x1]);
  dst[i] = (1.f - fy) * ((1.f - fx) * v00 + fx * v01) + fy * ((1.f - fx) * v10 + fx * v11);
}

// ---------------- host orchestration ----------------
extern "C" void kernel_launch(void* const* d_in, const int* in_sizes, int n_in,
                              void* d_out, int out_size, void* d_ws, size_t ws_size,
                              hipStream_t stream) {
  const float* a3     = (const float*)d_in[0];
  const float* a4     = (const float*)d_in[1];
  const float* conv_w = (const float*)d_in[2];
  const float* conv_b = (const float*)d_in[3];
  const float* bn_g   = (const float*)d_in[4];
  const float* bn_b   = (const float*)d_in[5];
  const float* q_w    = (const float*)d_in[6];
  const float* q_b    = (const float*)d_in[7];
  const float* q_ln_g = (const float*)d_in[8];
  const float* q_ln_b = (const float*)d_in[9];
  const float* k_w    = (const float*)d_in[10];
  const float* k_b    = (const float*)d_in[11];
  const float* k_ln_g = (const float*)d_in[12];
  const float* k_ln_b = (const float*)d_in[13];
  const float* v_w    = (const float*)d_in[14];
  const float* v_b    = (const float*)d_in[15];
  const float* out_w  = (const float*)d_in[16];
  const float* out_b  = (const float*)d_in[17];
  const float* lin1_w = (const float*)d_in[18];
  const float* lin1_b = (const float*)d_in[19];
  const float* lin2_w = (const float*)d_in[20];
  const float* lin2_b = (const float*)d_in[21];
  const float* n1_g   = (const float*)d_in[22];
  const float* n1_b   = (const float*)d_in[23];
  const float* n2_g   = (const float*)d_in[24];
  const float* n2_b   = (const float*)d_in[25];
  // d_in[26] == idx, known to be 10 (this implements the idx==10 path)
  (void)in_sizes; (void)n_in; (void)out_size; (void)ws_size;

  char* w = (char*)d_ws;
  auto carve = [&](size_t bytes) -> char* {
    char* p = w; w += (bytes + 255) & ~(size_t)255; return p;
  };
  float*  pos   = (float*)carve((size_t)N4 * C1 * 4);
  float*  resid = (float*)carve((size_t)TTOK * C1 * 4);
  __bf16* a3q   = (__bf16*)carve((size_t)TTOK * C1 * 2);
  __bf16* a4tok = (__bf16*)carve((size_t)TTOK * C2 * 2);
  __bf16* a4f   = (__bf16*)carve((size_t)TTOK * C1 * 2);
  __bf16* wconv = (__bf16*)carve((size_t)C1 * C2 * 2);
  __bf16* wq    = (__bf16*)carve((size_t)C1 * C1 * 2);
  __bf16* wk    = (__bf16*)carve((size_t)C1 * C1 * 2);
  __bf16* wv    = (__bf16*)carve((size_t)C1 * C1 * 2);
  __bf16* wo    = (__bf16*)carve((size_t)C1 * C1 * 2);
  __bf16* wl1   = (__bf16*)carve((size_t)4 * C1 * C1 * 2);
  __bf16* wl2   = (__bf16*)carve((size_t)4 * C1 * C1 * 2);
  float*  Qb    = (float*)carve((size_t)TTOK * C1 * 4);
  float*  Kb    = (float*)carve((size_t)TTOK * C1 * 4);  // Kb,Vb contiguous (aliased by ffh)
  float*  Vb    = (float*)carve((size_t)TTOK * C1 * 4);
  __bf16* attnb = (__bf16*)carve((size_t)TTOK * C1 * 2);
  // aliases (lifetimes disjoint along the stream order):
  float*  oraw = Qb;               // out-proj raw / then x (post norm1+residual)
  __bf16* xbf  = a3q;              // bf16 copy of x for FFN
  __bf16* ffh  = (__bf16*)Kb;      // 25600x1024 bf16 = 52.4MB over Kb+Vb
  float*  ff2  = (float*)a4tok;    // FFN raw out / then final tokens (out2)

  const int TB = 256;
  // pos table, weight converts, input prep
  hipLaunchKernelGGL(k_pos, dim3((N4 * C1) / TB), dim3(TB), 0, stream, pos);
  hipLaunchKernelGGL(k_cvt, dim3((C1 * C2 + TB - 1) / TB), dim3(TB), 0, stream, conv_w, wconv, C1 * C2);
  hipLaunchKernelGGL(k_cvt, dim3((C1 * C1 + TB - 1) / TB), dim3(TB), 0, stream, q_w, wq, C1 * C1);
  hipLaunchKernelGGL(k_cvt, dim3((C1 * C1 + TB - 1) / TB), dim3(TB), 0, stream, k_w, wk, C1 * C1);
  hipLaunchKernelGGL(k_cvt, dim3((C1 * C1 + TB - 1) / TB), dim3(TB), 0, stream, v_w, wv, C1 * C1);
  hipLaunchKernelGGL(k_cvt, dim3((C1 * C1 + TB - 1) / TB), dim3(TB), 0, stream, out_w, wo, C1 * C1);
  hipLaunchKernelGGL(k_cvt, dim3((4 * C1 * C1 + TB - 1) / TB), dim3(TB), 0, stream, lin1_w, wl1, 4 * C1 * C1);
  hipLaunchKernelGGL(k_cvt, dim3((4 * C1 * C1 + TB - 1) / TB), dim3(TB), 0, stream, lin2_w, wl2, 4 * C1 * C1);
  hipLaunchKernelGGL(k_prep_a3, dim3((BSZ * C1 * N4) / TB), dim3(TB), 0, stream, a3, pos, resid, a3q);
  hipLaunchKernelGGL(k_a4tok, dim3(N4 / 32, C2 / 32, BSZ), dim3(TB), 0, stream, a4, a4tok);

  // conv(1x1)+BN+pos -> a4f (bf16);   waves = (M/16)*(N/64)
  hipLaunchKernelGGL((k_gemm<1>), dim3((TTOK / 16) * (C1 / 64) / 8), dim3(TB), 0, stream,
                     a4tok, wconv, conv_b, (float*)nullptr, a4f, pos, bn_g, bn_b, TTOK, C1, C2);
  // Q/K/V projections (fp32 raw)
  hipLaunchKernelGGL((k_gemm<0>), dim3((TTOK / 16) * (C1 / 64) / 8), dim3(TB), 0, stream,
                     a3q, wq, q_b, Qb, (__bf16*)nullptr, pos, nullptr, nullptr, TTOK, C1, C1);
  hipLaunchKernelGGL((k_gemm<0>), dim3((TTOK / 16) * (C1 / 64) / 8), dim3(TB), 0, stream,
                     a4f, wk, k_b, Kb, (__bf16*)nullptr, pos, nullptr, nullptr, TTOK, C1, C1);
  hipLaunchKernelGGL((k_gemm<0>), dim3((TTOK / 16) * (C1 / 64) / 8), dim3(TB), 0, stream,
                     a4f, wv, v_b, Vb, (__bf16*)nullptr, pos, nullptr, nullptr, TTOK, C1, C1);
  // LN+relu on Q,K (in-place)
  hipLaunchKernelGGL(k_ln, dim3(TTOK * 32 / TB), dim3(TB), 0, stream, Qb, q_ln_g, q_ln_b,
                     (const float*)nullptr, Qb, (__bf16*)nullptr, 0);
  hipLaunchKernelGGL(k_ln, dim3(TTOK * 32 / TB), dim3(TB), 0, stream, Kb, k_ln_g, k_ln_b,
                     (const float*)nullptr, Kb, (__bf16*)nullptr, 0);
  // cosFormer attention
  hipLaunchKernelGGL(k_attn, dim3(BSZ * NHEAD), dim3(TB), 0, stream, Qb, Kb, Vb, attnb);
  // out projection -> oraw; norm1 + residual -> x (fp32 in oraw, bf16 in xbf)
  hipLaunchKernelGGL((k_gemm<0>), dim3((TTOK / 16) * (C1 / 64) / 8), dim3(TB), 0, stream,
                     attnb, wo, out_b, oraw, (__bf16*)nullptr, pos, nullptr, nullptr, TTOK, C1, C1);
  hipLaunchKernelGGL(k_ln, dim3(TTOK * 32 / TB), dim3(TB), 0, stream, oraw, n1_g, n1_b,
                     resid, oraw, xbf, 1);
  // FFN: lin1+gelu (bf16) -> lin2 -> norm2 + residual -> out2 (in ff2)
  hipLaunchKernelGGL((k_gemm<3>), dim3((TTOK / 16) * (4 * C1 / 64) / 8), dim3(TB), 0, stream,
                     xbf, wl1, lin1_b, (float*)nullptr, ffh, pos, nullptr, nullptr, TTOK, 4 * C1, C1);
  hipLaunchKernelGGL((k_gemm<0>), dim3((TTOK / 16) * (C1 / 64) / 8), dim3(TB), 0, stream,
                     ffh, wl2, lin2_b, ff2, (__bf16*)nullptr, pos, nullptr, nullptr, TTOK, C1, 4 * C1);
  hipLaunchKernelGGL(k_ln, dim3(TTOK * 32 / TB), dim3(TB), 0, stream, ff2, n2_g, n2_b,
                     oraw, ff2, (__bf16*)nullptr, 2);
  // outputs
  float* out0 = (float*)d_out;
  float* out1 = out0 + (size_t)BSZ * C1 * HW3;
  hipLaunchKernelGGL(k_up_a3, dim3((BSZ * C1 * HW3) / TB), dim3(TB), 0, stream, ff2, a3, out0);
  hipLaunchKernelGGL(k_up_a4, dim3((BSZ * C2 * HW3) / TB), dim3(TB), 0, stream, a4, out1);
}
